// CrossAttention_21449066676460
// MI455X (gfx1250) — matrix operand order repaired and checked by reading, not compiled, not run on hardware
//
#include <hip/hip_runtime.h>
#include <hip/hip_bf16.h>
#include <stdint.h>

// Problem constants (from reference)
#define DMODEL 1024
#define NHEADS 16
#define DHEAD  64
#define BATCH  4
#define QLEN   1024
#define TLEN   4096

// ---- CDNA5 WMMA types -------------------------------------------------------
typedef __attribute__((ext_vector_type(16))) __bf16   v16bf;
typedef __attribute__((ext_vector_type(8)))  float    v8f;
typedef __attribute__((ext_vector_type(4)))  uint32_t v4u;
typedef __attribute__((ext_vector_type(8)))  uint32_t v8u;
typedef __attribute__((ext_vector_type(4)))  int      v4i;

#if defined(__has_builtin)
#if __has_builtin(__builtin_amdgcn_global_load_async_to_lds_b128) && \
    __has_builtin(__builtin_amdgcn_s_wait_asynccnt)
#define ASYNC_LDS 1
#endif
#endif

#ifdef ASYNC_LDS
// async-DMA pointer types: b128 payload = int4; global = AS(1), LDS = AS(3)
typedef __attribute__((address_space(1))) v4i g_v4i;
typedef __attribute__((address_space(3))) v4i l_v4i;
#endif

__device__ __forceinline__ unsigned short f2bf(float x) {
  __hip_bfloat16 h = __float2bfloat16(x);
  return __builtin_bit_cast(unsigned short, h);
}

// 16x32 bf16 A-matrix fragment. Layout used: lane L -> row M = L%16;
// halves 0..7 = K in [16*half, 16*half+8), halves 8..15 = K + 8 (contiguous).
__device__ __forceinline__ v16bf ldA(const unsigned short* p_lo,
                                     const unsigned short* p_hi) {
  v4u lo = *(const v4u*)p_lo;
  v4u hi = *(const v4u*)p_hi;
  v8u r;
  r[0] = lo[0]; r[1] = lo[1]; r[2] = lo[2]; r[3] = lo[3];
  r[4] = hi[0]; r[5] = hi[1]; r[6] = hi[2]; r[7] = hi[3];
  return __builtin_bit_cast(v16bf, r);
}

// 32x16 bf16 B-matrix fragment: lane L holds column N = L%16 with
// K = [16*(L/16), 16*(L/16)+16) contiguous -> one 32B load.
__device__ __forceinline__ v16bf ldB(const unsigned short* p) {
  v8u r = *(const v8u*)p;
  return __builtin_bit_cast(v16bf, r);
}

__device__ __forceinline__ v8f wmma_bf16(v16bf a, v16bf b, v8f c) {
  return __builtin_amdgcn_wmma_f32_16x16x32_bf16(
      false, a, false, b, (short)0, c, false, false);
}

// VALU cross-lane max over 16-lane groups via v_permlane16_b32 (no DS pipe).
__device__ __forceinline__ float pl16(float v, unsigned s0, unsigned s1) {
  unsigned u = __builtin_bit_cast(unsigned, v);
  unsigned r = __builtin_amdgcn_permlane16(u, u, s0, s1, false, false);
  return __builtin_bit_cast(float, r);
}
__device__ __forceinline__ float rowmax16(float x) {
  x = fmaxf(x, pl16(x, 0x67452301u, 0x67452301u));  // xor 1
  x = fmaxf(x, pl16(x, 0x54761032u, 0x54761032u));  // xor 2
  x = fmaxf(x, pl16(x, 0x32107654u, 0x32107654u));  // xor 4
  x = fmaxf(x, pl16(x, 0xfedcba98u, 0x76543210u));  // xor 8
  return x;
}

// Stage 64B (32 bf16) of the A chunk: async DMA if available, else via VGPRs.
__device__ __forceinline__ void stage64(const unsigned short* src,
                                        unsigned short* dst) {
#ifdef ASYNC_LDS
  g_v4i* g = (g_v4i*)src;
  l_v4i* l = (l_v4i*)dst;
  __builtin_amdgcn_global_load_async_to_lds_b128(g,     l,     0, 0);
  __builtin_amdgcn_global_load_async_to_lds_b128(g + 1, l + 1, 0, 0);
  __builtin_amdgcn_global_load_async_to_lds_b128(g + 2, l + 2, 0, 0);
  __builtin_amdgcn_global_load_async_to_lds_b128(g + 3, l + 3, 0, 0);
#else
  v8u* d = (v8u*)dst;
  d[0] = *(const v8u*)src;
  d[1] = *(const v8u*)(src + 16);
#endif
}

// ---- Kernel 1: f32 -> bf16 cast --------------------------------------------
__global__ void cast_f32_bf16(const float* __restrict__ in,
                              unsigned short* __restrict__ out, int n) {
  int i = blockIdx.x * blockDim.x + threadIdx.x;
  int stride = gridDim.x * blockDim.x;
  for (; i < n; i += stride) out[i] = f2bf(in[i]);
}

// ---- Kernel 2: Y = X @ W^T (bf16 WMMA, f32 accumulate) ---------------------
// Block = 256 threads = 8 waves; block tile M=32, N=256; wave tile 16x64.
// A chunks double-buffered in LDS via async global->LDS DMA: fetch of chunk
// k+1 overlaps the 32 WMMAs of chunk k.
// MODE 0: bf16 head-split [B,H,seq,DHEAD]   MODE 1: bf16 [B,H,DHEAD,seq]
// MODE 2: f32 row-major [M, DMODEL]. seq = 1<<seq_shift.
template <int MODE>
__global__ void gemm_xwT(const unsigned short* __restrict__ X,
                         const unsigned short* __restrict__ W,
                         unsigned short* __restrict__ Yb,
                         float* __restrict__ Yf, int seq_shift) {
  __shared__ alignas(64) unsigned short As[2][32 * 256];  // 2 x 16 KB

  const int K    = DMODEL;
  const int lane = threadIdx.x & 31;
  const int wave = threadIdx.x >> 5;
  const int half = lane >> 4;
  const int ln   = lane & 15;
  const int msub = wave >> 2;   // 0..1
  const int nsub = wave & 3;    // 0..3

  const int m0 = blockIdx.x * 32 + msub * 16;
  const int n0 = blockIdx.y * 256 + nsub * 64;

  const unsigned short* w0 = W + (size_t)(n0 + ln) * K;  // B col n0+ln
  const unsigned short* Ablk = X + (size_t)(blockIdx.x * 32) * K;

  // cooperative A-load mapping: 8 threads/row, 32 elems (64B) each
  const int arow = threadIdx.x >> 3;
  const int acol = (threadIdx.x & 7) * 32;
  const unsigned short* asrc = Ablk + (size_t)arow * K + acol;

  v8f acc[4] = {v8f{}, v8f{}, v8f{}, v8f{}};

  int cur = 0;
  stage64(asrc, &As[0][arow * 256 + acol]);  // prologue: chunk 0

  for (int kc = 0; kc < K; kc += 256) {
#ifdef ASYNC_LDS
    __builtin_amdgcn_s_wait_asynccnt(0);
#endif
    __syncthreads();  // buf[cur] ready; prior reads of buf[cur^1] done

    if (kc + 256 < K)  // overlap next chunk's fetch with this chunk's WMMAs
      stage64(asrc + kc + 256, &As[cur ^ 1][arow * 256 + acol]);

#pragma unroll
    for (int ks = 0; ks < 256; ks += 32) {
      const unsigned short* ar = &As[cur][(msub * 16 + ln) * 256 + ks];
      const v16bf a = ldA(ar + 16 * half, ar + 16 * half + 8);
      const int kb = kc + ks + 16 * half;
#pragma unroll
      for (int j = 0; j < 4; ++j) {
        const v16bf b = ldB(w0 + (size_t)(j * 16) * K + kb);
        acc[j] = wmma_bf16(a, b, acc[j]);
      }
    }
    cur ^= 1;
  }

  // store: C/D layout -> lane 0-15: (row r, col ln); lane 16-31: row r+8
#pragma unroll
  for (int j = 0; j < 4; ++j) {
#pragma unroll
    for (int r = 0; r < 8; ++r) {
      const int m = m0 + r + 8 * half;
      const int n = n0 + j * 16 + ln;
      const float v = acc[j][r];
      if (MODE == 2) {
        Yf[(size_t)m * DMODEL + n] = v;
      } else {
        const int b_ = m >> seq_shift;
        const int s  = m & ((1 << seq_shift) - 1);
        const int h = n >> 6, d = n & 63;
        size_t off;
        if (MODE == 0)
          off = ((((size_t)(b_ * NHEADS + h)) << seq_shift) + s) * DHEAD + d;
        else
          off = ((((size_t)(b_ * NHEADS + h)) * DHEAD + d) << seq_shift) + s;
        Yb[off] = f2bf(v);
      }
    }
  }
}

// ---- Kernel 3: flash attention ---------------------------------------------
// Qh: [BH,QLEN,64] bf16, Kh: [BH,TLEN,64] bf16, Vt: [BH,64,TLEN] bf16,
// O: [B*QLEN, DMODEL] bf16. One wave: 16 q rows x 64 d; T tiled by 32.
__global__ void flash_attn(const unsigned short* __restrict__ Qh,
                           const unsigned short* __restrict__ Kh,
                           const unsigned short* __restrict__ Vt,
                           unsigned short* __restrict__ O) {
  __shared__ alignas(32) unsigned short ldsP[8][16 * 32];  // per-wave P tile

  const int lane = threadIdx.x & 31;
  const int wave = threadIdx.x >> 5;
  const int half = lane >> 4;
  const int ln   = lane & 15;

  const int qtiles = QLEN / 16;
  const int tile   = blockIdx.x * 8 + wave;
  const int bh     = tile / qtiles;
  const int qt     = tile % qtiles;
  const int b      = bh / NHEADS;
  const int h      = bh % NHEADS;

  const unsigned short* qbase =
      Qh + (((size_t)bh * QLEN) + qt * 16 + ln) * DHEAD;
  const v16bf qa0 = ldA(qbase + 16 * half, qbase + 8 + 16 * half);
  const v16bf qa1 = ldA(qbase + 32 + 16 * half, qbase + 40 + 16 * half);

  // constant all-1.0 bf16 B fragment: wmma(P, ones) -> per-row sums of P
  // replicated across lanes, directly in C-layout.
  v16bf ones;
  {
    v8u u;
#pragma unroll
    for (int i = 0; i < 8; ++i) u[i] = 0x3f803f80u;
    ones = __builtin_bit_cast(v16bf, u);
  }

  v8f o0 = {}, o1 = {}, o2 = {}, o3 = {};
  float mrow[8], lrow[8];
#pragma unroll
  for (int r = 0; r < 8; ++r) { mrow[r] = -3.0e38f; lrow[r] = 0.0f; }

  const float scale = 0.125f;       // 1/sqrt(DHEAD)
  const float log2e = 1.44269504f;

  const unsigned short* kbh = Kh + (size_t)bh * TLEN * DHEAD;
  const unsigned short* vbh = Vt + (size_t)bh * DHEAD * TLEN;
  unsigned short* myP = &ldsP[wave][0];

  for (int t0 = 0; t0 < TLEN; t0 += 32) {
    // ---- S = Q K^T for 32 key columns -> two 16x16 f32 tiles
    v8f s0 = {}, s1 = {};
    {
      const unsigned short* k0p = kbh + (size_t)(t0 + ln) * DHEAD;
      const unsigned short* k1p = kbh + (size_t)(t0 + 16 + ln) * DHEAD;
      s0 = wmma_bf16(qa0, ldB(k0p + 16 * half), s0);
      s0 = wmma_bf16(qa1, ldB(k0p + 32 + 16 * half), s0);
      s1 = wmma_bf16(qa0, ldB(k1p + 16 * half), s1);
      s1 = wmma_bf16(qa1, ldB(k1p + 32 + 16 * half), s1);
    }

    // ---- online softmax. Row max: pure-VALU permlane16 tree (no DS pipe).
    float sa[8], sc[8], corr[8];
#pragma unroll
    for (int r = 0; r < 8; ++r) {
      sa[r] = s0[r] * scale;
      sc[r] = s1[r] * scale;
      const float mnew = fmaxf(mrow[r], rowmax16(fmaxf(sa[r], sc[r])));
      corr[r] = exp2f((mrow[r] - mnew) * log2e);
      mrow[r] = mnew;
      sa[r] = exp2f((sa[r] - mnew) * log2e);
      sc[r] = exp2f((sc[r] - mnew) * log2e);
      o0[r] *= corr[r]; o1[r] *= corr[r];
      o2[r] *= corr[r]; o3[r] *= corr[r];
      const int prow = r + 8 * half;
      myP[prow * 32 + ln]      = f2bf(sa[r]);
      myP[prow * 32 + 16 + ln] = f2bf(sc[r]);
    }

    // ---- reload P as a 16x32 A fragment (same-wave DS ordering)
    const unsigned short* pr = myP + ln * 32 + 16 * half;
    const v16bf pa = ldA(pr, pr + 8);

    // ---- row sums of P via the matrix unit: wmma(P, ones) gives the
    //      16-col-replicated row sum in C-layout (kills the sum shuffles).
    {
      v8f ps = {};
      ps = wmma_bf16(pa, ones, ps);
#pragma unroll
      for (int r = 0; r < 8; ++r) lrow[r] = lrow[r] * corr[r] + ps[r];
    }

    // ---- O += P @ V (V transposed: column d contiguous in t)
    {
      const size_t tb = (size_t)t0 + 16 * half;
      o0 = wmma_bf16(pa, ldB(vbh + (size_t)(ln) * TLEN + tb), o0);
      o1 = wmma_bf16(pa, ldB(vbh + (size_t)(16 + ln) * TLEN + tb), o1);
      o2 = wmma_bf16(pa, ldB(vbh + (size_t)(32 + ln) * TLEN + tb), o2);
      o3 = wmma_bf16(pa, ldB(vbh + (size_t)(48 + ln) * TLEN + tb), o3);
    }
  }

  // ---- normalize and store pre-projection output [B*QLEN, DMODEL] bf16
#pragma unroll
  for (int r = 0; r < 8; ++r) {
    const float inv = 1.0f / lrow[r];
    const int qrow = qt * 16 + r + 8 * half;
    unsigned short* orow =
        O + ((size_t)(b * QLEN + qrow)) * DMODEL + h * DHEAD + ln;
    orow[0]  = f2bf(o0[r] * inv);
    orow[16] = f2bf(o1[r] * inv);
    orow[32] = f2bf(o2[r] * inv);
    orow[48] = f2bf(o3[r] * inv);
  }
}

// ---- Host-side launcher -----------------------------------------------------
extern "C" void kernel_launch(void* const* d_in, const int* in_sizes, int n_in,
                              void* d_out, int out_size, void* d_ws,
                              size_t ws_size, hipStream_t stream) {
  const float* q_in  = (const float*)d_in[0];
  const float* kv_in = (const float*)d_in[1];
  const float* Wq    = (const float*)d_in[2];
  const float* Wk    = (const float*)d_in[3];
  const float* Wv    = (const float*)d_in[4];
  const float* Wo    = (const float*)d_in[5];
  float* out = (float*)d_out;

  char* ws = (char*)d_ws;
  size_t off = 0;
  auto carve = [&](size_t bytes) -> void* {
    void* p = ws + off;
    off += (bytes + 255) & ~(size_t)255;
    return p;
  };
  const size_t NQ = (size_t)BATCH * QLEN * DMODEL;
  const size_t NT = (size_t)BATCH * TLEN * DMODEL;
  const size_t NW = (size_t)DMODEL * DMODEL;
  unsigned short* qb   = (unsigned short*)carve(NQ * 2);
  unsigned short* kvb  = (unsigned short*)carve(NT * 2);
  unsigned short* wqb  = (unsigned short*)carve(NW * 2);
  unsigned short* wkb  = (unsigned short*)carve(NW * 2);
  unsigned short* wvb  = (unsigned short*)carve(NW * 2);
  unsigned short* wob  = (unsigned short*)carve(NW * 2);
  unsigned short* Qh   = (unsigned short*)carve(NQ * 2);
  unsigned short* Kh   = (unsigned short*)carve(NT * 2);
  unsigned short* Vt   = (unsigned short*)carve(NT * 2);
  unsigned short* attn = (unsigned short*)carve(NQ * 2);
  (void)ws_size; (void)in_sizes; (void)n_in; (void)out_size;

  cast_f32_bf16<<<4096, 256, 0, stream>>>(q_in,  qb,  (int)NQ);
  cast_f32_bf16<<<4096, 256, 0, stream>>>(kv_in, kvb, (int)NT);
  cast_f32_bf16<<<1024, 256, 0, stream>>>(Wq, wqb, (int)NW);
  cast_f32_bf16<<<1024, 256, 0, stream>>>(Wk, wkb, (int)NW);
  cast_f32_bf16<<<1024, 256, 0, stream>>>(Wv, wvb, (int)NW);
  cast_f32_bf16<<<1024, 256, 0, stream>>>(Wo, wob, (int)NW);

  dim3 blk(256);
  dim3 gq(BATCH * QLEN / 32, DMODEL / 256);   // (128, 4)
  dim3 gkv(BATCH * TLEN / 32, DMODEL / 256);  // (512, 4)
  gemm_xwT<0><<<gq,  blk, 0, stream>>>(qb,  wqb, Qh, nullptr, 10);
  gemm_xwT<0><<<gkv, blk, 0, stream>>>(kvb, wkb, Kh, nullptr, 12);
  gemm_xwT<1><<<gkv, blk, 0, stream>>>(kvb, wvb, Vt, nullptr, 12);

  flash_attn<<<(BATCH * NHEADS * (QLEN / 16)) / 8, blk, 0, stream>>>(
      Qh, Kh, Vt, attn);

  gemm_xwT<2><<<gq, blk, 0, stream>>>(attn, wob, nullptr, out, 10);
}
